// GroupAttention_26216480374957
// MI455X (gfx1250) — matrix operand order repaired and checked by reading, not compiled, not run on hardware
//
#include <hip/hip_runtime.h>
#include <math.h>

typedef _Float16 h8  __attribute__((ext_vector_type(8)));
typedef _Float16 h16 __attribute__((ext_vector_type(16)));
typedef float    f8  __attribute__((ext_vector_type(8)));
typedef int      v4i __attribute__((ext_vector_type(4)));

#define TOKS 15552      // B*N = 64*243
#define CTOT 2176       // 17*128
#define NSEQ 243
#define NPAD 256

// address-space-qualified int4 pointers for the async-DMA builtin
typedef __attribute__((address_space(1))) v4i* gv4p;   // global (AS1)
typedef __attribute__((address_space(3))) v4i* lv4p;   // LDS (AS3)

#if defined(__has_builtin)
# if __has_builtin(__builtin_amdgcn_global_load_async_to_lds_b128) && \
     __has_builtin(__builtin_amdgcn_s_wait_asynccnt)
#  define USE_ASYNC_LDS 1
# endif
#endif
#ifndef USE_ASYNC_LDS
# define USE_ASYNC_LDS 0
#endif

// GROUPS = ((1,2,3),(4,5,6),(0,7,8,9,10),(11,12,13),(14,15,16))
__device__ __constant__ int g_chunks[5][5] = {
  {1,2,3,0,0},{4,5,6,0,0},{0,7,8,9,10},{11,12,13,0,0},{14,15,16,0,0}};

static __device__ inline h16 cat16(h8 lo, h8 hi){
  return __builtin_shufflevector(lo,hi,0,1,2,3,4,5,6,7,8,9,10,11,12,13,14,15);
}

// A fragment (16x32 f16, M x K): lane holds row (lane&15); k-base 0/8 by lane half,
// chunks [kb..kb+7] and [16+kb..16+kb+7] -> two b128 loads.
static __device__ inline h16 load_frag_a(const _Float16* base, int lda, int mBase, int kt){
  int lane = threadIdx.x & 31;
  const _Float16* p = base + (size_t)(mBase + (lane & 15)) * lda + kt*32 + ((lane < 16) ? 0 : 8);
  return cat16(*(const h8*)p, *(const h8*)(p + 16));
}

// B fragment (32x16 f16, K x N) from row-major W (N x K): lane holds column (lane&15)
// = row of W; k-base 0/8 by lane half, chunks [kb..kb+7] and [16+kb..16+kb+7] -> two b128 loads.
static __device__ inline h16 load_frag_b(const _Float16* base, int ldb, int nBase, int kt){
  int lane = threadIdx.x & 31;
  const _Float16* p = base + (size_t)(nBase + (lane & 15)) * ldb + kt*32 + ((lane < 16) ? 0 : 8);
  return cat16(*(const h8*)p, *(const h8*)(p + 16));
}

static __device__ inline f8 wmma16(h16 a, h16 b, f8 c){
  return __builtin_amdgcn_wmma_f32_16x16x32_f16(false, a, false, b, (short)0, c, false, false);
}

// ---------------------------------------------------------------- fp32 -> fp16
__global__ void f32_to_f16_kernel(const float* __restrict__ s, _Float16* __restrict__ d, int n){
  for (int i = blockIdx.x*blockDim.x + threadIdx.x; i < n; i += gridDim.x*blockDim.x)
    d[i] = (_Float16)s[i];
}

// ---------------------------------------------------------------- block reduce
static __device__ inline float block_sum_128(float v){
  __shared__ float red[4];
  #pragma unroll
  for (int m = 16; m >= 1; m >>= 1) v += __shfl_xor(v, m, 32);
  __syncthreads();
  if ((threadIdx.x & 31) == 0) red[threadIdx.x >> 5] = v;
  __syncthreads();
  return red[0] + red[1] + red[2] + red[3];
}

// ------------------------------------------------- LN1 + channel-group gather
__global__ __launch_bounds__(128) void ln_gather_kernel(
    const float* __restrict__ x, const float* __restrict__ w, const float* __restrict__ b,
    int group, int C, float* __restrict__ xi, _Float16* __restrict__ h){
  int tok = blockIdx.x;
  int cnt = C >> 7;                       // elements per thread (3 or 5), blockDim=128
  float v[5];
  float s = 0.f;
  #pragma unroll
  for (int i = 0; i < 5; ++i) if (i < cnt){
    int c = threadIdx.x + (i << 7);
    v[i] = x[(size_t)tok*CTOT + g_chunks[group][c >> 7]*128 + (c & 127)];
    s += v[i];
  }
  float mean = block_sum_128(s) / (float)C;
  float sq = 0.f;
  #pragma unroll
  for (int i = 0; i < 5; ++i) if (i < cnt){ float d = v[i] - mean; sq += d*d; }
  float rstd = rsqrtf(block_sum_128(sq) / (float)C + 1e-5f);
  #pragma unroll
  for (int i = 0; i < 5; ++i) if (i < cnt){
    int c = threadIdx.x + (i << 7);
    xi[(size_t)tok*C + c] = v[i];
    h [(size_t)tok*C + c] = (_Float16)((v[i] - mean)*rstd*w[c] + b[c]);
  }
}

// ------------------------------------------------- LN2 (contiguous input)
__global__ __launch_bounds__(128) void ln_kernel(
    const float* __restrict__ xin, const float* __restrict__ w, const float* __restrict__ b,
    int C, _Float16* __restrict__ h){
  int tok = blockIdx.x;
  int cnt = C >> 7;
  float v[5];
  float s = 0.f;
  #pragma unroll
  for (int i = 0; i < 5; ++i) if (i < cnt){
    v[i] = xin[(size_t)tok*C + threadIdx.x + (i << 7)];
    s += v[i];
  }
  float mean = block_sum_128(s) / (float)C;
  float sq = 0.f;
  #pragma unroll
  for (int i = 0; i < 5; ++i) if (i < cnt){ float d = v[i] - mean; sq += d*d; }
  float rstd = rsqrtf(block_sum_128(sq) / (float)C + 1e-5f);
  #pragma unroll
  for (int i = 0; i < 5; ++i) if (i < cnt){
    int c = threadIdx.x + (i << 7);
    h[(size_t)tok*C + c] = (_Float16)((v[i] - mean)*rstd*w[c] + b[c]);
  }
}

// ------------------------------------------------- WMMA GEMM: Y = A(MxK) * W(NxK)^T
// Block = 4 waves = 64(M) x 128(N) tile. A tile (64x32 f16) staged to LDS once per
// K-step (async DMA to LDS when available), double-buffered; waves read A fragments
// from LDS (ds_load_b128) and B fragments per-wave from global (L2-resident weights).
// EPI 0: -> f16          (qkv)
// EPI 1: +bias +resid -> f32       (proj residual)
// EPI 2: +bias, exact GELU -> f16  (fc1)
// EPI 3: +bias +resid -> f32 scattered into d_out channel layout (fc2)
template<int EPI>
__global__ __launch_bounds__(128) void gemm_kernel(
    const _Float16* __restrict__ A, const _Float16* __restrict__ W, int K, int Nout,
    _Float16* __restrict__ o16, float* __restrict__ o32,
    const float* __restrict__ bias, const float* __restrict__ resid,
    float* __restrict__ oscat, int group){
  __shared__ _Float16 As[2][64*32];
  int wv   = threadIdx.x >> 5;
  int lane = threadIdx.x & 31;
  int tid  = threadIdx.x;
  int nBase = blockIdx.x*128 + wv*32;
  int mBase = blockIdx.y*64;
  int kTiles = K >> 5;

  // Stage one 64x32 f16 A tile into LDS: 256 x 16B chunks, 2 per thread.
  auto stage = [&](int buf, int kt){
    #pragma unroll
    for (int i = 0; i < 2; ++i){
      int c   = tid + i*128;        // chunk id
      int row = c >> 2, q = c & 3;  // 4 x 8-half chunks per row
      const _Float16* g = A + (size_t)(mBase + row)*K + kt*32 + q*8;
      _Float16* l = &As[buf][row*32 + q*8];
#if USE_ASYNC_LDS
      __builtin_amdgcn_global_load_async_to_lds_b128((gv4p)(void*)g, (lv4p)l, 0, 0);
#else
      *(h8*)l = *(const h8*)g;
#endif
    }
  };

  f8 acc[4][2] = {};
  stage(0, 0);
  for (int kt = 0; kt < kTiles; ++kt){
    int cur = kt & 1;
#if USE_ASYNC_LDS
    __builtin_amdgcn_s_wait_asynccnt(0);
#endif
    __syncthreads();                       // staged tile visible; prev reads done
    if (kt + 1 < kTiles) stage(cur ^ 1, kt + 1);   // overlap next DMA with math
    h16 b0 = load_frag_b(W, K, nBase,      kt);
    h16 b1 = load_frag_b(W, K, nBase + 16, kt);
    if (kt + 1 < kTiles)
      __builtin_prefetch((const void*)(W + (size_t)(nBase + (lane & 15))*K + (kt + 1)*32), 0, 3);
    h16 a[4];
    #pragma unroll
    for (int mi = 0; mi < 4; ++mi){
      const _Float16* lp = &As[cur][(mi*16 + (lane & 15))*32 + ((lane < 16) ? 0 : 8)];
      a[mi] = cat16(*(const h8*)lp, *(const h8*)(lp + 16));
    }
    #pragma unroll
    for (int mi = 0; mi < 4; ++mi){
      acc[mi][0] = wmma16(a[mi], b0, acc[mi][0]);
      acc[mi][1] = wmma16(a[mi], b1, acc[mi][1]);
    }
  }

  int half = lane >> 4, cil = lane & 15;
  #pragma unroll
  for (int ni = 0; ni < 2; ++ni){
    int n = nBase + ni*16 + cil;
    float bv = (EPI != 0) ? bias[n] : 0.f;
    #pragma unroll
    for (int mi = 0; mi < 4; ++mi){
      #pragma unroll
      for (int r = 0; r < 8; ++r){
        int m = mBase + mi*16 + r + half*8;
        float va = acc[mi][ni][r];
        if (EPI == 0){
          o16[(size_t)m*Nout + n] = (_Float16)va;
        } else if (EPI == 1){
          o32[(size_t)m*Nout + n] = va + bv + resid[(size_t)m*Nout + n];
        } else if (EPI == 2){
          float t = va + bv;
          o16[(size_t)m*Nout + n] = (_Float16)(0.5f*t*(1.f + erff(t*0.70710678118f)));
        } else {
          float t = va + bv + resid[(size_t)m*Nout + n];
          oscat[(size_t)m*CTOT + g_chunks[group][n >> 7]*128 + (n & 127)] = t;
        }
      }
    }
  }
}

// ------------------------------------------------- qkv -> padded Q, K, V^T
__global__ void qkv_split_kernel(const _Float16* __restrict__ qkv, int C, int hd, int hdp,
                                 float qscale, _Float16* __restrict__ Q,
                                 _Float16* __restrict__ Kb, _Float16* __restrict__ Vt){
  int idx = blockIdx.x*blockDim.x + threadIdx.x;
  int total = 512*NPAD*hdp;
  if (idx >= total) return;
  int dp = idx % hdp; int t = idx / hdp; int n = t % NPAD; int bh = t / NPAD;
  _Float16 qv = (_Float16)0.f, kv = qv, vv = qv;
  if (n < NSEQ && dp < hd){
    int bb = bh >> 3, hh = bh & 7;
    size_t base = (size_t)(bb*NSEQ + n)*(size_t)(3*C) + hh*hd + dp;
    qv = (_Float16)((float)qkv[base] * qscale);
    kv = qkv[base + C];
    vv = qkv[base + 2*C];
  }
  size_t rm = (size_t)bh*NPAD*hdp + (size_t)n*hdp + dp;
  Q[rm]  = qv;
  Kb[rm] = kv;
  Vt[(size_t)bh*NPAD*hdp + (size_t)dp*NPAD + n] = vv;   // (hd x tokens)
}

// ------------------------------------------------- flash attention, wave = 16 q rows
template<int HD, int HDP>
__global__ __launch_bounds__(256) void attn_kernel(
    const _Float16* __restrict__ Q, const _Float16* __restrict__ Kb,
    const _Float16* __restrict__ Vt, _Float16* __restrict__ aout, int C){
  constexpr int NF = HD/16;     // output hd fragments (3 or 5)
  constexpr int KK = HDP/32;    // QK^T k-steps (2 or 3)
  __shared__ _Float16 pbuf[8][16*32];
  int wv = threadIdx.x >> 5, lane = threadIdx.x & 31;
  int half = lane >> 4, cil = lane & 15;
  int bh = blockIdx.x >> 1;
  int qRow0 = (blockIdx.x & 1)*128 + wv*16;
  int bb = bh >> 3, hh = bh & 7;
  const _Float16* Qp = Q  + (size_t)bh*NPAD*HDP;
  const _Float16* Kp = Kb + (size_t)bh*NPAD*HDP;
  const _Float16* Vp = Vt + (size_t)bh*NPAD*HDP;
  h16 qf[KK];
  #pragma unroll
  for (int kk = 0; kk < KK; ++kk) qf[kk] = load_frag_a(Qp, HDP, qRow0, kk);
  f8 o[NF] = {};
  float mrow[8], lrow[8];
  #pragma unroll
  for (int r = 0; r < 8; ++r){ mrow[r] = -3e38f; lrow[r] = 0.f; }
  _Float16* pb = pbuf[wv];

  for (int ct = 0; ct < NPAD/32; ++ct){
    int colBase = ct*32;
    f8 s0 = {}, s1 = {};
    #pragma unroll
    for (int kk = 0; kk < KK; ++kk){
      h16 k0 = load_frag_b(Kp, HDP, colBase,      kk);
      h16 k1 = load_frag_b(Kp, HDP, colBase + 16, kk);
      s0 = wmma16(qf[kk], k0, s0);
      s1 = wmma16(qf[kk], k1, s1);
    }
    bool msk0 = (colBase + cil)      >= NSEQ;   // lane owns one key column per frag
    bool msk1 = (colBase + 16 + cil) >= NSEQ;
    float tmax[8];
    #pragma unroll
    for (int r = 0; r < 8; ++r){
      if (msk0) s0[r] = -3e38f;
      if (msk1) s1[r] = -3e38f;
      tmax[r] = fmaxf(s0[r], s1[r]);
    }
    #pragma unroll
    for (int mk = 8; mk >= 1; mk >>= 1)
      #pragma unroll
      for (int r = 0; r < 8; ++r) tmax[r] = fmaxf(tmax[r], __shfl_xor(tmax[r], mk, 32));
    float p0[8], p1[8], psum[8], alpha[8];
    #pragma unroll
    for (int r = 0; r < 8; ++r){
      float mn = fmaxf(mrow[r], tmax[r]);
      alpha[r] = expf(mrow[r] - mn);
      p0[r] = expf(s0[r] - mn);
      p1[r] = expf(s1[r] - mn);
      psum[r] = p0[r] + p1[r];
      mrow[r] = mn;
      lrow[r] *= alpha[r];
    }
    #pragma unroll
    for (int mk = 8; mk >= 1; mk >>= 1)
      #pragma unroll
      for (int r = 0; r < 8; ++r) psum[r] += __shfl_xor(psum[r], mk, 32);
    #pragma unroll
    for (int r = 0; r < 8; ++r) lrow[r] += psum[r];
    #pragma unroll
    for (int nf = 0; nf < NF; ++nf)
      #pragma unroll
      for (int r = 0; r < 8; ++r) o[nf][r] *= alpha[r];

    // C-layout -> A-layout relayout of P through per-wave LDS (16x32 f16 tile)
    #pragma unroll
    for (int r = 0; r < 8; ++r){
      pb[(r + half*8)*32 + cil]      = (_Float16)p0[r];
      pb[(r + half*8)*32 + 16 + cil] = (_Float16)p1[r];
    }
    __syncthreads();
    h16 pa;
    { const _Float16* pp = pb + (size_t)cil*32 + ((lane < 16) ? 0 : 8);
      pa = cat16(*(const h8*)pp, *(const h8*)(pp + 16)); }
    #pragma unroll
    for (int nf = 0; nf < NF; ++nf){
      const _Float16* vp = Vp + (size_t)(nf*16 + cil)*NPAD + colBase + ((lane < 16) ? 0 : 8);
      h16 vb = cat16(*(const h8*)vp, *(const h8*)(vp + 16));
      o[nf] = wmma16(pa, vb, o[nf]);
    }
    __syncthreads();
  }
  #pragma unroll
  for (int nf = 0; nf < NF; ++nf){
    int ch = hh*HD + nf*16 + cil;
    #pragma unroll
    for (int r = 0; r < 8; ++r){
      int row = qRow0 + r + half*8;
      if (row < NSEQ)
        aout[(size_t)(bb*NSEQ + row)*C + ch] = (_Float16)(o[nf][r] / lrow[r]);
    }
  }
}

// ================================================================ host
extern "C" void kernel_launch(void* const* d_in, const int* in_sizes, int n_in,
                              void* d_out, int out_size, void* d_ws, size_t ws_size,
                              hipStream_t stream){
  (void)in_sizes; (void)n_in; (void)out_size; (void)ws_size;
  const float* x = (const float*)d_in[0];
  float* out = (float*)d_out;
  char* ws = (char*)d_ws;
  const int lens[5] = {3,3,5,3,3};

  size_t off = 0;
  auto carve = [&](size_t bytes){ size_t o = off; off += (bytes + 255) & ~(size_t)255; return o; };
  const size_t w16_off = carve((size_t)12*640*640*2);         // fp16 weights (per group, reused)
  const size_t xi_off  = carve((size_t)TOKS*640*4);           // fp32 residual / x2
  const size_t h_off   = carve((size_t)TOKS*640*2);           // fp16 LN output
  const size_t ao_off  = carve((size_t)TOKS*640*2);           // fp16 attention output
  const size_t qkv_sz  = (size_t)TOKS*3*640*2;
  const size_t qkvpad  = (size_t)512*NPAD*96*2;               // one of Q/K/Vt, max size
  const size_t big_off = carve(qkv_sz + 3*qkvpad);            // qkv16 + Q + K + Vt ; m1 aliases

  for (int g = 0; g < 5; ++g){
    const int C = 128*lens[g], hd = 16*lens[g];
    const int hdp = (hd + 31)/32*32;                          // 64 or 96
    const float qscale = 1.f / sqrtf((float)hd);
    const float* const* pin = (const float* const*)(d_in + 1 + g*11);
    const float *n1w = pin[0], *n1b = pin[1], *qkvw = pin[2], *projw = pin[3], *projb = pin[4],
                *n2w = pin[5], *n2b = pin[6], *fc1w = pin[7], *fc1b = pin[8],
                *fc2w = pin[9], *fc2b = pin[10];

    _Float16* qkvw16 = (_Float16*)(ws + w16_off);
    _Float16* projw16 = qkvw16 + (size_t)3*C*C;
    _Float16* fc1w16  = projw16 + (size_t)C*C;
    _Float16* fc2w16  = fc1w16 + (size_t)4*C*C;
    float*    xi    = (float*)(ws + xi_off);
    _Float16* hbuf  = (_Float16*)(ws + h_off);
    _Float16* aoutb = (_Float16*)(ws + ao_off);
    _Float16* qkv16 = (_Float16*)(ws + big_off);
    _Float16* Qb    = (_Float16*)(ws + big_off + qkv_sz);
    _Float16* Kbf   = (_Float16*)((char*)Qb + qkvpad);
    _Float16* Vt    = (_Float16*)((char*)Kbf + qkvpad);
    _Float16* m1    = qkv16;   // alias: qkv/Q/K/Vt dead by fc1 time

    f32_to_f16_kernel<<<dim3(512), dim3(256), 0, stream>>>(qkvw,  qkvw16, 3*C*C);
    f32_to_f16_kernel<<<dim3(512), dim3(256), 0, stream>>>(projw, projw16, C*C);
    f32_to_f16_kernel<<<dim3(512), dim3(256), 0, stream>>>(fc1w,  fc1w16, 4*C*C);
    f32_to_f16_kernel<<<dim3(512), dim3(256), 0, stream>>>(fc2w,  fc2w16, 4*C*C);

    ln_gather_kernel<<<dim3(TOKS), dim3(128), 0, stream>>>(x, n1w, n1b, g, C, xi, hbuf);

    gemm_kernel<0><<<dim3(3*C/128, TOKS/64), dim3(128), 0, stream>>>(
        hbuf, qkvw16, C, 3*C, qkv16, nullptr, nullptr, nullptr, nullptr, g);

    { int total = 512*NPAD*hdp;
      qkv_split_kernel<<<dim3((total + 255)/256), dim3(256), 0, stream>>>(
          qkv16, C, hd, hdp, qscale, Qb, Kbf, Vt); }

    if (g == 2) attn_kernel<80,96><<<dim3(1024), dim3(256), 0, stream>>>(Qb, Kbf, Vt, aoutb, C);
    else        attn_kernel<48,64><<<dim3(1024), dim3(256), 0, stream>>>(Qb, Kbf, Vt, aoutb, C);

    gemm_kernel<1><<<dim3(C/128, TOKS/64), dim3(128), 0, stream>>>(
        aoutb, projw16, C, C, nullptr, xi, projb, xi, nullptr, g);   // x2 = xi + proj(o)

    ln_kernel<<<dim3(TOKS), dim3(128), 0, stream>>>(xi, n2w, n2b, C, hbuf);

    gemm_kernel<2><<<dim3(4*C/128, TOKS/64), dim3(128), 0, stream>>>(
        hbuf, fc1w16, C, 4*C, m1, nullptr, fc1b, nullptr, nullptr, g);

    gemm_kernel<3><<<dim3(C/128, TOKS/64), dim3(128), 0, stream>>>(
        m1, fc2w16, 4*C, C, nullptr, nullptr, fc2b, xi, out, g);     // scatter to d_out
  }
}
